// TreeLayer_53541062312192
// MI455X (gfx1250) — compile-verified, matrix operand order checked by reading, not run on hardware
//
#include <hip/hip_runtime.h>

// ---------------- problem constants (fixed by reference) ----------------
#define HEADS 4
#define N0_ 32768
#define N1_ 8192
#define N2_ 2048
#define D0_ 64
#define D1_ 128
#define D2_ 256
#define KNN_ 16
#define PCB_ 4          // point-cloud batches
#define SCALE_ 8.0f     // sqrt(OUT_PLANES)=sqrt(64)

typedef __attribute__((ext_vector_type(16))) _Float16 v16h;
typedef __attribute__((ext_vector_type(4)))  _Float16 v4h;
typedef __attribute__((ext_vector_type(8)))  float    v8f;
typedef __attribute__((ext_vector_type(4)))  unsigned v4u;
typedef __attribute__((ext_vector_type(8)))  int      v8i;
typedef __attribute__((ext_vector_type(4)))  int      v4i;

// =====================================================================
// WMMA GEMM (direct path): C[N,Do] = A[N,Dk] * W[Dk,Do]
// f32 in/out, f16 MACs, f32 acc. 128 thr = 4 waves, one 16x16 tile/wave.
// b128 global loads + packed b64 LDS stores (no per-element wait chains).
// =====================================================================
__global__ __launch_bounds__(128) void gemm_wmma(
    const float* __restrict__ A, const float* __restrict__ W,
    float* __restrict__ C, int N, int Dk, int Do)
{
  __shared__ _Float16 As[64][36];   // 64 x 32K, stride 36 halves (8B-aligned rows)
  __shared__ _Float16 Ws[16][34];   // W^T tile: 16 cols x 32K
  const int t    = threadIdx.x;
  const int wave = t >> 5;
  const int lane = t & 31;
  const int rowBase = blockIdx.x * 64;
  const int colBase = blockIdx.y * 16;
  const int mr  = lane & 15;          // A row within tile / C column
  const int hi  = (lane >> 4) << 3;   // 0/8: K offset for frags, M offset for C

  v8f acc = {};
  for (int k0 = 0; k0 < Dk; k0 += 32) {
    // A tile 64x32: 512 float4 chunks, 4 per thread, fully vectorized
#pragma unroll
    for (int i = 0; i < 4; ++i) {
      int e  = t + i * 128;
      int r  = e >> 3;                // 8 chunks per 32-wide row
      int c4 = (e & 7) << 2;
      const float4 v = *(const float4*)&A[(size_t)(rowBase + r) * Dk + (k0 + c4)];
      v4h hv;
      hv[0] = (_Float16)v.x; hv[1] = (_Float16)v.y;
      hv[2] = (_Float16)v.z; hv[3] = (_Float16)v.w;
      *(v4h*)&As[r][c4] = hv;
    }
    // W tile 32x16 -> transposed LDS: one float4 (4 cols) per thread
    {
      int c  = t >> 2;                // K row 0..31
      int n4 = (t & 3) << 2;          // col group
      const float4 w = *(const float4*)&W[(size_t)(k0 + c) * Do + (colBase + n4)];
      Ws[n4 + 0][c] = (_Float16)w.x;
      Ws[n4 + 1][c] = (_Float16)w.y;
      Ws[n4 + 2][c] = (_Float16)w.z;
      Ws[n4 + 3][c] = (_Float16)w.w;
    }
    if (k0 + 32 < Dk)                 // pull next A tile toward the caches
      __builtin_prefetch(&A[(size_t)(rowBase + (t >> 1)) * Dk + (k0 + 32)], 0, 1);
    __syncthreads();

    // fragments per ISA 7.12.2 16-bit layouts (B mirrored by column)
    v16h af, bf;
#pragma unroll
    for (int h = 0; h < 16; ++h) {
      int kk = ((h >> 3) << 4) + hi + (h & 7);
      af[h] = As[wave * 16 + mr][kk];
      bf[h] = Ws[mr][kk];
    }
    acc = __builtin_amdgcn_wmma_f32_16x16x32_f16(false, af, false, bf,
                                                 (short)0, acc, false, false);
    __syncthreads();
  }
#pragma unroll
  for (int r = 0; r < 8; ++r) {       // C: VGPR r -> M = r + hi, N = mr
    int row = rowBase + wave * 16 + hi + r;
    C[(size_t)row * Do + (colBase + mr)] = acc[r];
  }
}

// =====================================================================
// TDM descriptor helper: 2D tile load Global->LDS (data_size = 4B).
// D# packing per CDNA5 ISA:
//   g0: [count=1 | lds_addr | global_addr(57b) | type=2]
//   g1: [data_size=2(4B) | tensor_dim0/1 | tile_dim0/1 | dim0_stride]
// =====================================================================
__device__ __forceinline__ void tdm_load_2d(
    unsigned ldsOff, const void* gptr,
    unsigned tensorW, unsigned tensorH, unsigned strideW,
    unsigned tileW, unsigned tileH)
{
  unsigned long long ga = (unsigned long long)gptr;
  v4u g0;
  g0[0] = 1u;                                        // count=1, user mode
  g0[1] = ldsOff;                                    // LDS byte address
  g0[2] = (unsigned)(ga & 0xFFFFFFFFu);              // global_addr[31:0]
  g0[3] = (unsigned)((ga >> 32) & 0x1FFFFFFu)        // global_addr[56:32]
        | (2u << 30);                                // type = 2 ("image")
  v8i g1;
  g1[0] = (int)(2u << 16);                           // data_size = 4B; no mask/pad
  g1[1] = (int)((tensorW & 0xFFFFu) << 16);          // tensor_dim0[15:0]
  g1[2] = (int)((tensorW >> 16) | ((tensorH & 0xFFFFu) << 16)); // dim0 hi | dim1 lo
  g1[3] = (int)((tensorH >> 16) | (tileW << 16));    // dim1 hi | tile_dim0
  g1[4] = (int)tileH;                                // tile_dim1 (tile_dim2 = 0)
  g1[5] = (int)strideW;                              // tensor_dim0_stride[31:0]
  g1[6] = 0;                                         // stride hi | dim1_stride lo
  g1[7] = 0;
  v4i gz = {0, 0, 0, 0};
#if defined(__clang_major__) && (__clang_major__ >= 23)
  v8i gz8 = {0, 0, 0, 0, 0, 0, 0, 0};
  __builtin_amdgcn_tensor_load_to_lds(g0, g1, gz, gz, gz8, 0);
#else
  __builtin_amdgcn_tensor_load_to_lds(g0, g1, gz, gz, 0);
#endif
}

// =====================================================================
// WMMA GEMM (Tensor Data Mover path): tiles DMA'd Global->LDS as f32 by
// TENSOR_LOAD_TO_LDS, converted to f16 during fragment build.
// =====================================================================
__global__ __launch_bounds__(128) void gemm_wmma_tdm(
    const float* __restrict__ A, const float* __restrict__ W,
    float* __restrict__ C, int N, int Dk, int Do)
{
  __shared__ float Asf[64 * 32];      // A tile, row-major, contiguous
  __shared__ float Bsf[32 * 16];      // W tile, K-major (as in memory)
  const int t    = threadIdx.x;
  const int wave = t >> 5;
  const int lane = t & 31;
  const int rowBase = blockIdx.x * 64;
  const int colBase = blockIdx.y * 16;
  const int mr  = lane & 15;
  const int hi  = (lane >> 4) << 3;
  const unsigned ldsA = (unsigned)(size_t)(void*)Asf;  // LDS offset (addr[31:0])
  const unsigned ldsB = (unsigned)(size_t)(void*)Bsf;

  v8f acc = {};
  for (int k0 = 0; k0 < Dk; k0 += 32) {
    if (t < 32) {                      // one wave drives the DMA
      tdm_load_2d(ldsA, &A[(size_t)rowBase * Dk + k0], (unsigned)Dk, (unsigned)N,
                  (unsigned)Dk, 32u, 64u);
      tdm_load_2d(ldsB, &W[(size_t)k0 * Do + colBase], (unsigned)Do, (unsigned)Dk,
                  (unsigned)Do, 16u, 32u);
      __builtin_amdgcn_s_wait_tensorcnt(0);
    }
    __syncthreads();                   // publish DMA'd tiles to all waves

    v16h af, bf;
#pragma unroll
    for (int h = 0; h < 16; ++h) {
      int kk = ((h >> 3) << 4) + hi + (h & 7);
      af[h] = (_Float16)Asf[(wave * 16 + mr) * 32 + kk];
      bf[h] = (_Float16)Bsf[kk * 16 + mr];
    }
    acc = __builtin_amdgcn_wmma_f32_16x16x32_f16(false, af, false, bf,
                                                 (short)0, acc, false, false);
    __syncthreads();
  }
#pragma unroll
  for (int r = 0; r < 8; ++r) {
    int row = rowBase + wave * 16 + hi + r;
    C[(size_t)row * Do + (colBase + mr)] = acc[r];
  }
}

// =====================================================================
// KNN (k=16) within each equal-sized batch. One thread per query,
// LDS tile of 256 candidates, fully unrolled register top-16.
// =====================================================================
__global__ __launch_bounds__(256) void knn_kernel(
    const float* __restrict__ p, int n, int* __restrict__ outIdx)
{
  __shared__ float sx[256], sy[256], sz[256];
  const int q    = blockIdx.x * 256 + threadIdx.x;
  const int base = (q / n) * n;
  const float qx = p[q * 3 + 0], qy = p[q * 3 + 1], qz = p[q * 3 + 2];
  float bd[16]; int bi[16];
#pragma unroll
  for (int s = 0; s < 16; ++s) { bd[s] = 3.4e38f; bi[s] = 0; }

  for (int tile = 0; tile < n; tile += 256) {
    int c = base + tile + threadIdx.x;
    sx[threadIdx.x] = p[c * 3 + 0];
    sy[threadIdx.x] = p[c * 3 + 1];
    sz[threadIdx.x] = p[c * 3 + 2];
    __syncthreads();
    for (int jj = 0; jj < 256; ++jj) {
      float dx = qx - sx[jj], dy = qy - sy[jj], dz = qz - sz[jj];
      float d = dx * dx + dy * dy + dz * dz;
      if (d < bd[15]) {
        bd[15] = d; bi[15] = tile + jj;
#pragma unroll
        for (int s = 15; s > 0; --s) {            // bubble into place
          if (bd[s] < bd[s - 1]) {
            float td = bd[s - 1]; bd[s - 1] = bd[s]; bd[s] = td;
            int   ti = bi[s - 1]; bi[s - 1] = bi[s]; bi[s] = ti;
          }
        }
      }
    }
    __syncthreads();
  }
#pragma unroll
  for (int s = 0; s < 16; ++s) outIdx[(size_t)q * 16 + s] = bi[s];
}

// =====================================================================
// pe-MLP: Linear(3,3) -> BatchNorm(global edge stats) -> ReLU -> Linear(3,dh)
// =====================================================================
struct PeParams {
  const float *W1, *b1, *g, *be, *W2, *b2;
  const float* stats;  // [0..2]=sum(h), [3..5]=sum(h^2)
  float invE;
  int dh;
};

__device__ __forceinline__ void pe_hidden(const PeParams& P,
                                          float px, float py, float pz, float r[3])
{
#pragma unroll
  for (int c = 0; c < 3; ++c) {
    float h  = px * P.W1[c] + py * P.W1[3 + c] + pz * P.W1[6 + c] + P.b1[c];
    float mu = P.stats[c] * P.invE;
    float vr = P.stats[3 + c] * P.invE - mu * mu;
    float hn = (h - mu) * rsqrtf(vr + 1e-5f) * P.g[c] + P.be[c];
    r[c] = fmaxf(hn, 0.f);
  }
}
__device__ __forceinline__ float pe_out(const PeParams& P, const float r[3], int j)
{
  return r[0] * P.W2[j] + r[1] * P.W2[P.dh + j] + r[2] * P.W2[2 * P.dh + j] + P.b2[j];
}

// ---- stats pass: sum(h), sum(h^2) per channel over all edges of a module ----
// mode 0: parent edge (q=e, k=e>>2); 1: child edge (q=e>>2, k=e); 2: sibling KNN
__global__ __launch_bounds__(256) void pe_stats_kernel(
    const float* __restrict__ pq, const float* __restrict__ pk,
    const int* __restrict__ nbr, int mode, int E, int n,
    const float* __restrict__ W1, const float* __restrict__ b1,
    float* __restrict__ stats)
{
  __shared__ float acc[6];
  if (threadIdx.x < 6) acc[threadIdx.x] = 0.f;
  __syncthreads();
  int e = blockIdx.x * 256 + threadIdx.x;
  if (e < E) {
    int qi, ki;
    if (mode == 0)      { qi = e;       ki = e >> 2; }
    else if (mode == 1) { qi = e >> 2;  ki = e; }
    else { int node = e >> 4; qi = node; ki = (node / n) * n + nbr[e]; }
    float px = pq[qi * 3 + 0] - pk[ki * 3 + 0];
    float py = pq[qi * 3 + 1] - pk[ki * 3 + 1];
    float pz = pq[qi * 3 + 2] - pk[ki * 3 + 2];
#pragma unroll
    for (int c = 0; c < 3; ++c) {
      float h = px * W1[c] + py * W1[3 + c] + pz * W1[6 + c] + b1[c];
      atomicAdd(&acc[c], h);          // ds_add_f32
      atomicAdd(&acc[3 + c], h * h);
    }
  }
  __syncthreads();
  if (threadIdx.x < 6) atomicAdd(&stats[threadIdx.x], acc[threadIdx.x]);
}

// =====================================================================
// Parents: each query has exactly one edge -> softmax weight == 1:
//   out[q, h*Dh+j] = Vs[parent, h*Dh+j] + pe_j
// =====================================================================
__global__ __launch_bounds__(256) void parents_kernel(
    const float* __restrict__ pq, const float* __restrict__ pk,
    const float* __restrict__ Vs, PeParams P, int D, float* __restrict__ outp)
{
  int gid = blockIdx.x * 256 + threadIdx.x;
  int q = gid / D, d = gid % D;
  int j = d % P.dh;
  int par = q >> 2;
  float px = pq[q * 3 + 0] - pk[par * 3 + 0];
  float py = pq[q * 3 + 1] - pk[par * 3 + 1];
  float pz = pq[q * 3 + 2] - pk[par * 3 + 2];
  float r3[3]; pe_hidden(P, px, py, pz, r3);
  outp[gid] = Vs[(size_t)par * D + d] + pe_out(P, r3, j);
}

// =====================================================================
// Children (EDGES=4, keys=4q+e) / Siblings (EDGES=16, keys from KNN idx).
// One block per query (blockDim = feature dim); segment softmax per head.
// =====================================================================
template <int DT, int EDGES>
__global__ __launch_bounds__(DT) void local_attn_kernel(
    const float* __restrict__ pq, const float* __restrict__ pk,
    const float* __restrict__ Qm, const float* __restrict__ Km,
    const float* __restrict__ Vm, const int* __restrict__ nbr,
    int n, PeParams P, float denom, float* __restrict__ outp)
{
  constexpr int Dh = DT / HEADS;
  __shared__ float red[DT];
  __shared__ float wsh[EDGES * HEADS];
  const int q = blockIdx.x;
  const int d = threadIdx.x;
  const int h = d / Dh;
  const int j = d % Dh;
  const float qv = Qm[(size_t)q * DT + d];
  const float qx = pq[q * 3 + 0], qy = pq[q * 3 + 1], qz = pq[q * 3 + 2];

  float pev[EDGES]; int keyv[EDGES];
#pragma unroll
  for (int e = 0; e < EDGES; ++e) {
    int key = nbr ? (q / n) * n + nbr[(size_t)q * EDGES + e] : q * 4 + e;
    keyv[e] = key;
    float px = qx - pk[key * 3 + 0];
    float py = qy - pk[key * 3 + 1];
    float pz = qz - pk[key * 3 + 2];
    float r3[3]; pe_hidden(P, px, py, pz, r3);
    pev[e] = pe_out(P, r3, j);
    red[d] = Km[(size_t)key * DT + d] * qv + pev[e];  // k*q + pe, summed over j
#pragma unroll
    for (int off = Dh >> 1; off > 0; off >>= 1) {
      __syncthreads();
      if (j < off) red[d] += red[d + off];
    }
    __syncthreads();
    if (j == 0) wsh[e * HEADS + h] = red[d] / denom;
    __syncthreads();
  }
  float wm = -3.4e38f;
#pragma unroll
  for (int e = 0; e < EDGES; ++e) wm = fmaxf(wm, wsh[e * HEADS + h]);
  float av[EDGES]; float s = 0.f;
#pragma unroll
  for (int e = 0; e < EDGES; ++e) { av[e] = expf(wsh[e * HEADS + h] - wm); s += av[e]; }
  const float inv = 1.0f / s;
  float o = 0.f;
#pragma unroll
  for (int e = 0; e < EDGES; ++e)
    o += av[e] * inv * (Vm[(size_t)keyv[e] * DT + d] + pev[e]);
  outp[(size_t)q * DT + d] = o;
}

// ---- out = softmax(w3)[0]*parents + [1]*children + [2]*siblings ----
__global__ __launch_bounds__(256) void combine_kernel(
    const float* __restrict__ par, const float* __restrict__ chi,
    const float* __restrict__ sib, const float* __restrict__ w3,
    int total, float* __restrict__ outp)
{
  int gid = blockIdx.x * 256 + threadIdx.x;
  if (gid >= total) return;
  float w0 = w3[0], w1 = w3[1], w2 = w3[2];
  float m  = fmaxf(w0, fmaxf(w1, w2));
  float e0 = expf(w0 - m), e1 = expf(w1 - m), e2 = expf(w2 - m);
  float is = 1.0f / (e0 + e1 + e2);
  float pv = par ? par[gid] : 0.f;
  float cv = chi ? chi[gid] : 0.f;
  float sv = sib ? sib[gid] : 0.f;
  outp[gid] = (e0 * pv + e1 * cv + e2 * sv) * is;
}

// =====================================================================
// host orchestration
// =====================================================================
static inline PeParams mkPe(void* const* d_in, int base, const float* stats,
                            float invE, int dh)
{
  PeParams P;
  P.W1 = (const float*)d_in[base + 0];
  P.b1 = (const float*)d_in[base + 1];
  P.g  = (const float*)d_in[base + 2];
  P.be = (const float*)d_in[base + 3];
  P.W2 = (const float*)d_in[base + 4];
  P.b2 = (const float*)d_in[base + 5];
  P.stats = stats; P.invE = invE; P.dh = dh;
  return P;
}

extern "C" void kernel_launch(void* const* d_in, const int* in_sizes, int n_in,
                              void* d_out, int out_size, void* d_ws, size_t ws_size,
                              hipStream_t stream)
{
  // inputs (setup_inputs dict order; params flattened in insertion order)
  const float* x0 = (const float*)d_in[0];
  const float* x1 = (const float*)d_in[1];
  const float* x2 = (const float*)d_in[2];
  const float* p0 = (const float*)d_in[3];
  const float* p1 = (const float*)d_in[4];
  const float* p2 = (const float*)d_in[5];
  const float* WQ[3] = {(const float*)d_in[10], (const float*)d_in[11], (const float*)d_in[12]};
  const float* WK[3] = {(const float*)d_in[13], (const float*)d_in[14], (const float*)d_in[15]};
  const float* WV[3] = {(const float*)d_in[16], (const float*)d_in[17], (const float*)d_in[18]};
  const float* Wek[2] = {(const float*)d_in[19], (const float*)d_in[20]};
  const float* Wev[2] = {(const float*)d_in[21], (const float*)d_in[22]};
  const float* Wsk[2] = {(const float*)d_in[23], (const float*)d_in[24]};
  const float* Wsv[2] = {(const float*)d_in[25], (const float*)d_in[26]};
  const int PE_PAR = 27, PE_CHI = 45, PE_SIB = 63;
  const float* w3 = (const float*)d_in[81];

  float* ws = (float*)d_ws;
  size_t off = 0;
  auto alloc = [&](size_t nf) { size_t o = off; off += (nf + 63) & ~(size_t)63; return o; };

  float* Q0  = ws + alloc((size_t)N0_ * D0_);
  float* K0f = ws + alloc((size_t)N0_ * D0_);
  float* V0f = ws + alloc((size_t)N0_ * D0_);
  float* Q1  = ws + alloc((size_t)N1_ * D1_);
  float* K1f = ws + alloc((size_t)N1_ * D1_);
  float* V1f = ws + alloc((size_t)N1_ * D1_);
  float* Q2  = ws + alloc((size_t)N2_ * D2_);
  float* K2f = ws + alloc((size_t)N2_ * D2_);
  float* V2f = ws + alloc((size_t)N2_ * D2_);
  float* Ks0 = ws + alloc((size_t)N1_ * D0_);
  float* Vs0 = ws + alloc((size_t)N1_ * D0_);
  float* Ks1 = ws + alloc((size_t)N2_ * D1_);
  float* Vs1 = ws + alloc((size_t)N2_ * D1_);
  float* Ke0 = ws + alloc((size_t)N0_ * D1_);
  float* Ve0 = ws + alloc((size_t)N0_ * D1_);
  float* Ke1 = ws + alloc((size_t)N1_ * D2_);
  float* Ve1 = ws + alloc((size_t)N1_ * D2_);
  float* par0 = ws + alloc((size_t)N0_ * D0_);
  float* par1 = ws + alloc((size_t)N1_ * D1_);
  float* chi1 = ws + alloc((size_t)N1_ * D1_);
  float* chi2 = ws + alloc((size_t)N2_ * D2_);
  float* sib0 = ws + alloc((size_t)N0_ * D0_);
  float* sib1 = ws + alloc((size_t)N1_ * D1_);
  int*   idx0 = (int*)(ws + alloc((size_t)N0_ * KNN_));
  int*   idx1 = (int*)(ws + alloc((size_t)N1_ * KNN_));
  float* stats = ws + alloc(64);     // 6 modules x 8 floats
  if (ws_size < off * sizeof(float)) return;

  auto gemm = [&](const float* A, const float* W, float* C, int N, int Dk, int Do) {
    gemm_wmma<<<dim3(N / 64, Do / 16), dim3(128), 0, stream>>>(A, W, C, N, Dk, Do);
  };
  auto gemmT = [&](const float* A, const float* W, float* C, int N, int Dk, int Do) {
    gemm_wmma_tdm<<<dim3(N / 64, Do / 16), dim3(128), 0, stream>>>(A, W, C, N, Dk, Do);
  };

  // ---- phase 1: projections (WMMA GEMMs) ----
  gemm(x0, WQ[0], Q0,  N0_, D0_, D0_);
  gemm(x0, WK[0], K0f, N0_, D0_, D0_);
  gemm(x0, WV[0], V0f, N0_, D0_, D0_);
  gemm(x1, WQ[1], Q1,  N1_, D1_, D1_);
  gemm(x1, WK[1], K1f, N1_, D1_, D1_);
  gemm(x1, WV[1], V1f, N1_, D1_, D1_);
  gemm(x2, WQ[2], Q2,  N2_, D2_, D2_);
  gemm(x2, WK[2], K2f, N2_, D2_, D2_);
  gemm(x2, WV[2], V2f, N2_, D2_, D2_);
  gemm(K1f, Wsk[0], Ks0, N1_, D1_, D0_);
  gemm(V1f, Wsv[0], Vs0, N1_, D1_, D0_);
  gemm(K2f, Wsk[1], Ks1, N2_, D2_, D1_);
  gemm(V2f, Wsv[1], Vs1, N2_, D2_, D1_);
  // largest projections via the Tensor Data Mover path
  gemmT(K0f, Wek[0], Ke0, N0_, D0_, D1_);
  gemmT(V0f, Wev[0], Ve0, N0_, D0_, D1_);
  gemmT(K1f, Wek[1], Ke1, N1_, D1_, D2_);
  gemmT(V1f, Wev[1], Ve1, N1_, D1_, D2_);

  // ---- phase 2: per-batch KNN ----
  knn_kernel<<<N0_ / 256, 256, 0, stream>>>(p0, N0_ / PCB_, idx0);
  knn_kernel<<<N1_ / 256, 256, 0, stream>>>(p1, N1_ / PCB_, idx1);

  // ---- phase 3: global BatchNorm stats for each pe-MLP ----
  hipMemsetAsync(stats, 0, 48 * sizeof(float), stream);
  pe_stats_kernel<<<N0_ / 256, 256, 0, stream>>>(p0, p1, nullptr, 0, N0_, 0,
      (const float*)d_in[PE_PAR + 0], (const float*)d_in[PE_PAR + 1], stats + 0);
  pe_stats_kernel<<<N1_ / 256, 256, 0, stream>>>(p1, p2, nullptr, 0, N1_, 0,
      (const float*)d_in[PE_PAR + 6], (const float*)d_in[PE_PAR + 7], stats + 8);
  pe_stats_kernel<<<N0_ / 256, 256, 0, stream>>>(p1, p0, nullptr, 1, N0_, 0,
      (const float*)d_in[PE_CHI + 6], (const float*)d_in[PE_CHI + 7], stats + 16);
  pe_stats_kernel<<<N1_ / 256, 256, 0, stream>>>(p2, p1, nullptr, 1, N1_, 0,
      (const float*)d_in[PE_CHI + 12], (const float*)d_in[PE_CHI + 13], stats + 24);
  pe_stats_kernel<<<(N0_ * KNN_) / 256, 256, 0, stream>>>(p0, p0, idx0, 2,
      N0_ * KNN_, N0_ / PCB_,
      (const float*)d_in[PE_SIB + 0], (const float*)d_in[PE_SIB + 1], stats + 32);
  pe_stats_kernel<<<(N1_ * KNN_) / 256, 256, 0, stream>>>(p1, p1, idx1, 2,
      N1_ * KNN_, N1_ / PCB_,
      (const float*)d_in[PE_SIB + 6], (const float*)d_in[PE_SIB + 7], stats + 40);

  // ---- phase 4: attention ----
  PeParams Ppar0 = mkPe(d_in, PE_PAR + 0,  stats + 0,  1.0f / N0_, D0_ / HEADS);
  PeParams Ppar1 = mkPe(d_in, PE_PAR + 6,  stats + 8,  1.0f / N1_, D1_ / HEADS);
  PeParams Pchi1 = mkPe(d_in, PE_CHI + 6,  stats + 16, 1.0f / N0_, D1_ / HEADS);
  PeParams Pchi2 = mkPe(d_in, PE_CHI + 12, stats + 24, 1.0f / N1_, D2_ / HEADS);
  PeParams Psib0 = mkPe(d_in, PE_SIB + 0,  stats + 32, 1.0f / (N0_ * KNN_), D0_ / HEADS);
  PeParams Psib1 = mkPe(d_in, PE_SIB + 6,  stats + 40, 1.0f / (N1_ * KNN_), D1_ / HEADS);

  parents_kernel<<<(N0_ * D0_) / 256, 256, 0, stream>>>(p0, p1, Vs0, Ppar0, D0_, par0);
  parents_kernel<<<(N1_ * D1_) / 256, 256, 0, stream>>>(p1, p2, Vs1, Ppar1, D1_, par1);

  const float dChild = SCALE_ * 4.0f;    // + 1e-8 vanishes in f32
  const float dSib   = SCALE_ * 16.0f;
  local_attn_kernel<D1_, 4><<<N1_, D1_, 0, stream>>>(p1, p0, Q1, Ke0, Ve0,
      nullptr, 0, Pchi1, dChild, chi1);
  local_attn_kernel<D2_, 4><<<N2_, D2_, 0, stream>>>(p2, p1, Q2, Ke1, Ve1,
      nullptr, 0, Pchi2, dChild, chi2);
  local_attn_kernel<D0_, 16><<<N0_, D0_, 0, stream>>>(p0, p0, Q0, K0f, V0f,
      idx0, N0_ / PCB_, Psib0, dSib, sib0);
  local_attn_kernel<D1_, 16><<<N1_, D1_, 0, stream>>>(p1, p1, Q1, K1f, V1f,
      idx1, N1_ / PCB_, Psib1, dSib, sib1);

  // ---- phase 5: weighted combine into d_out (out0|out1|out2 concatenated) ----
  float* out = (float*)d_out;
  combine_kernel<<<(N0_ * D0_) / 256, 256, 0, stream>>>(par0, nullptr, sib0, w3,
      N0_ * D0_, out);
  combine_kernel<<<(N1_ * D1_) / 256, 256, 0, stream>>>(par1, chi1, sib1, w3,
      N1_ * D1_, out + (size_t)N0_ * D0_);
  combine_kernel<<<(N2_ * D2_) / 256, 256, 0, stream>>>(nullptr, chi2, nullptr, w3,
      N2_ * D2_, out + (size_t)N0_ * D0_ + (size_t)N1_ * D1_);
}
